// Online_Quantizer_17995912970295
// MI455X (gfx1250) — matrix operand order repaired and checked by reading, not compiled, not run on hardware
//
#include <hip/hip_runtime.h>
#include <math.h>

#define N_VEC   16384   // B*H*W = 16*32*32
#define K_CODES 8192
#define C_DIM   256
#define BM 128          // rows per block tile
#define BN 128          // codes per block tile
#define CK 32           // k-chunk (WMMA K)

typedef __attribute__((ext_vector_type(16))) _Float16 v16h;
typedef __attribute__((ext_vector_type(8)))  _Float16 v8h;
typedef __attribute__((ext_vector_type(8)))  float    v8f;

#if defined(__has_builtin)
#  if __has_builtin(__builtin_amdgcn_global_load_async_to_lds_b128)
#    define HAVE_ASYNC_LDS 1
#  endif
#endif
#ifndef HAVE_ASYNC_LDS
#  define HAVE_ASYNC_LDS 0
#endif

#if HAVE_ASYNC_LDS
typedef int v4i_ __attribute__((ext_vector_type(4)));
typedef __attribute__((address_space(1))) v4i_ as1_v4i;   // global (AS1) int4
typedef __attribute__((address_space(3))) v4i_ as3_v4i;   // LDS (AS3) int4
#endif

// copy 16 bytes global -> LDS (async on gfx1250 if toolchain exposes it)
__device__ __forceinline__ void copy16_to_lds(const _Float16* g, _Float16* l) {
#if HAVE_ASYNC_LDS
    as1_v4i* gp = (as1_v4i*)(unsigned long long)(uintptr_t)g;
    as3_v4i* lp = (as3_v4i*)(unsigned)(uintptr_t)l;
    __builtin_amdgcn_global_load_async_to_lds_b128(gp, lp, 0, 0);
#else
    *reinterpret_cast<uint4*>(l) = *reinterpret_cast<const uint4*>(g);
#endif
}

__device__ __forceinline__ void wait_async_copies() {
#if HAVE_ASYNC_LDS
#  if __has_builtin(__builtin_amdgcn_s_wait_asynccnt)
    __builtin_amdgcn_s_wait_asynccnt(0);
#  else
    asm volatile("s_wait_asynccnt 0x0" ::: "memory");
#  endif
#endif
}

// order-preserving float -> u32 map (works for negatives too)
__device__ __forceinline__ unsigned fkey(float f) {
    unsigned u = __float_as_uint(f);
    unsigned mask = (unsigned)((int)u >> 31) | 0x80000000u;
    return u ^ mask;
}

__device__ __forceinline__ unsigned long long shflx64(unsigned long long v, int m) {
    unsigned lo = (unsigned)v, hi = (unsigned)(v >> 32);
    lo = __shfl_xor(lo, m, 32);
    hi = __shfl_xor(hi, m, 32);
    return (((unsigned long long)hi) << 32) | lo;
}

// ---------------- init ----------------
__global__ void vq_init(unsigned long long* rowmin, unsigned long long* colmax,
                        unsigned* hist) {
    int i = blockIdx.x * 256 + threadIdx.x;
    if (i < N_VEC) rowmin[i] = 0xFFFFFFFFFFFFFFFFull;
    if (i < K_CODES) { colmax[i] = 0ull; hist[i] = 0u; }
}

// ---------------- prep: z -> zf16 + row norms ----------------
__global__ void vq_prep_z(const float* __restrict__ z, _Float16* __restrict__ zf16,
                          float* __restrict__ znorm) {
    int n = blockIdx.x, c = threadIdx.x;
    int b = n >> 10, hw = n & 1023;
    float v = z[((size_t)(b * C_DIM + c) << 10) + hw];
    zf16[(size_t)n * C_DIM + c] = (_Float16)v;
    __shared__ float red[256];
    red[c] = v * v; __syncthreads();
    for (int s = 128; s > 0; s >>= 1) { if (c < s) red[c] += red[c + s]; __syncthreads(); }
    if (c == 0) znorm[n] = red[0];
}

// ---------------- prep: embedding -> ef16 + norms ----------------
__global__ void vq_prep_e(const float* __restrict__ emb, _Float16* __restrict__ ef16,
                          float* __restrict__ enorm) {
    int k = blockIdx.x, c = threadIdx.x;
    float v = emb[(size_t)k * C_DIM + c];
    ef16[(size_t)k * C_DIM + c] = (_Float16)v;
    __shared__ float red[256];
    red[c] = v * v; __syncthreads();
    for (int s = 128; s > 0; s >>= 1) { if (c < s) red[c] += red[c + s]; __syncthreads(); }
    if (c == 0) enorm[k] = red[0];
}

// ---------------- main WMMA distance kernel + fused argmin/argmax ----------------
__global__ __launch_bounds__(256) void vq_dist(
    const _Float16* __restrict__ zf16, const _Float16* __restrict__ ef16,
    const float* __restrict__ znorm, const float* __restrict__ enorm,
    unsigned long long* __restrict__ rowmin, unsigned long long* __restrict__ colmax)
{
    __shared__ __align__(32) _Float16 As[2][BM * CK];   // 2 x 8 KB (double buffered)
    __shared__ __align__(32) _Float16 Bs[2][BN * CK];   // 2 x 8 KB
    __shared__ unsigned long long cmax_s[BN];           // 1 KB

    const int tid  = threadIdx.x;
    const int lane = tid & 31;
    const int wave = tid >> 5;
    const int rowBase = blockIdx.y * BM;
    const int colBase = blockIdx.x * BN;

    if (tid < BN) cmax_s[tid] = 0ull;

    v8f acc[8] = {};

    const int m  = lane & 15;      // M (for A) / N (for B) within tile
    const int hs = lane >> 4;      // half-wave select
    const int k0 = hs << 3;        // A-fragment K offset: 0 or 8
    const int kb = hs << 4;        // unused; B fragment also uses k0

    // stage one 128x32 half tile pair (A rows / B codes) for element-chunk kc
    auto load_tiles = [&](int kc, int buf) {
        #pragma unroll
        for (int q = tid; q < 512; q += 256) {
            int row = q >> 2, part = q & 3;
            copy16_to_lds(zf16 + (size_t)(rowBase + row) * C_DIM + kc + part * 8,
                          &As[buf][row * CK + part * 8]);
        }
        #pragma unroll
        for (int q = tid; q < 512; q += 256) {
            int col = q >> 2, part = q & 3;
            copy16_to_lds(ef16 + (size_t)(colBase + col) * C_DIM + kc + part * 8,
                          &Bs[buf][col * CK + part * 8]);
        }
    };

    load_tiles(0, 0);
    wait_async_copies();
    __syncthreads();

    #pragma unroll
    for (int kci = 0; kci < 8; ++kci) {
        const int cur = kci & 1;
        if (kci < 7) load_tiles((kci + 1) * CK, cur ^ 1);  // prefetch next chunk

        // A fragment: ISA 16-bit 16x32 layout (lane half holds K {k0..k0+7, 16+k0..23+k0})
        const _Float16* arow = &As[cur][(wave * 16 + m) * CK];
        v8h alo = *reinterpret_cast<const v8h*>(arow + k0);
        v8h ahi = *reinterpret_cast<const v8h*>(arow + 16 + k0);
        v16h afrag;
        #pragma unroll
        for (int j = 0; j < 8; ++j) { afrag[j] = alo[j]; afrag[8 + j] = ahi[j]; }

        // preload ALL B fragments, then issue the 8 WMMAs back-to-back
        v16h bfrag[8];
        #pragma unroll
        for (int t = 0; t < 8; ++t) {
            const _Float16* brow = &Bs[cur][(t * 16 + m) * CK];
            v8h blo = *reinterpret_cast<const v8h*>(brow + k0);
            v8h bhi = *reinterpret_cast<const v8h*>(brow + 16 + k0);
            #pragma unroll
            for (int j = 0; j < 8; ++j) { bfrag[t][j] = blo[j]; bfrag[t][8 + j] = bhi[j]; }
        }

        #pragma unroll
        for (int t = 0; t < 8; ++t)
            acc[t] = __builtin_amdgcn_wmma_f32_16x16x32_f16(
                false, afrag, false, bfrag[t], (short)0, acc[t], false, false);

        if (kci < 7) {
            wait_async_copies();   // prefetched buffer complete (this wave)
            __syncthreads();       // publish to all waves; retire reads of old buffer
        }
    }

    // ---- epilogue: d = |z|^2 + |e|^2 - 2S, fused argmin(row)/argmax(col) ----
    const int growBase = rowBase + wave * 16 + hs * 8;  // D element r -> row growBase+r
    float zn[8];
    #pragma unroll
    for (int r = 0; r < 8; ++r) zn[r] = znorm[growBase + r];

    unsigned long long rbest[8];
    #pragma unroll
    for (int r = 0; r < 8; ++r) rbest[r] = 0xFFFFFFFFFFFFFFFFull;

    #pragma unroll
    for (int t = 0; t < 8; ++t) {
        int gcol = colBase + t * 16 + m;
        float en = enorm[gcol];
        unsigned long long cbest = 0ull;
        #pragma unroll
        for (int r = 0; r < 8; ++r) {
            float d = zn[r] + en - 2.0f * acc[t][r];
            unsigned fk = fkey(d);
            // min key: (dist, col)  -> ties pick smallest col (JAX first-index)
            unsigned long long kmin = (((unsigned long long)fk) << 32) | (unsigned)gcol;
            if (kmin < rbest[r]) rbest[r] = kmin;
            // max key: (dist, ~row) -> ties pick smallest row
            unsigned long long kmax = (((unsigned long long)fk) << 32) |
                                      (unsigned)(0xFFFFFFFFu - (unsigned)(growBase + r));
            if (kmax > cbest) cbest = kmax;
        }
        // combine the two half-wave rows for this column
        unsigned long long o = shflx64(cbest, 16);
        if (o > cbest) cbest = o;
        if (hs == 0) atomicMax(&cmax_s[t * 16 + m], cbest);
    }

    // per-row min across the 16 lanes of each half-wave (128 cols of this block)
    #pragma unroll
    for (int r = 0; r < 8; ++r) {
        #pragma unroll
        for (int msk = 1; msk < 16; msk <<= 1) {
            unsigned long long o = shflx64(rbest[r], msk);
            if (o < rbest[r]) rbest[r] = o;
        }
    }
    if (m == 0) {
        #pragma unroll
        for (int r = 0; r < 8; ++r) atomicMin(&rowmin[growBase + r], rbest[r]);
    }

    __syncthreads();
    if (tid < BN) atomicMax(&colmax[colBase + tid], cmax_s[tid]);
}

// ---------------- per-row finalize: token, histogram, z_q output, row sumsq ----------------
__global__ void vq_rows(const float* __restrict__ z, const float* __restrict__ emb,
                        const unsigned long long* __restrict__ rowmin,
                        unsigned* __restrict__ hist, float* __restrict__ rowsumsq,
                        float* __restrict__ out_zq) {
    int n = blockIdx.x, c = threadIdx.x;
    unsigned tok = (unsigned)(rowmin[n] & 0xFFFFFFFFull);
    if (c == 0) atomicAdd(&hist[tok], 1u);
    int b = n >> 10, hw = n & 1023;
    float e  = emb[(size_t)tok * C_DIM + c];
    float zv = z[((size_t)(b * C_DIM + c) << 10) + hw];
    out_zq[((size_t)(b * C_DIM + c) << 10) + hw] = e;   // straight-through == z_q numerically
    float d = e - zv;
    __shared__ float red[256];
    red[c] = d * d; __syncthreads();
    for (int s = 128; s > 0; s >>= 1) { if (c < s) red[c] += red[c + s]; __syncthreads(); }
    if (c == 0) rowsumsq[n] = red[0];
}

// ---------------- per-code finalize: EMA prob, anchor update ----------------
__global__ void vq_cols(const float* __restrict__ z, const float* __restrict__ emb,
                        const float* __restrict__ embprob,
                        const unsigned long long* __restrict__ colmax,
                        const unsigned* __restrict__ hist,
                        float* __restrict__ out_newemb, float* __restrict__ out_newprob) {
    int k = blockIdx.x, c = threadIdx.x;
    unsigned far_n = 0xFFFFFFFFu - (unsigned)(colmax[k] & 0xFFFFFFFFull);
    float avg = (float)hist[k] * (1.0f / (float)N_VEC);
    float np = embprob[k] * 0.99f + avg * 0.01f;
    if (c == 0) out_newprob[k] = np;
    float decay = expf(-(np * (float)K_CODES * 10.0f) / (1.0f - 0.99f) - 0.001f);
    int b = far_n >> 10, hw = far_n & 1023;
    float rf = z[((size_t)(b * C_DIM + c) << 10) + hw];
    float e  = emb[(size_t)k * C_DIM + c];
    out_newemb[(size_t)k * C_DIM + c] = e * (1.0f - decay) + rf * decay;
}

// ---------------- deterministic scalar reductions ----------------
__global__ void vq_scalars(const float* __restrict__ rowsumsq,
                           const unsigned* __restrict__ hist,
                           float* __restrict__ out4) {
    int t = threadIdx.x;
    __shared__ float red[256];
    float s = 0.f;
    for (int i = t; i < N_VEC; i += 256) s += rowsumsq[i];
    red[t] = s; __syncthreads();
    for (int k = 128; k > 0; k >>= 1) { if (t < k) red[t] += red[t + k]; __syncthreads(); }
    float total = red[0]; __syncthreads();

    float u = 0.f, H = 0.f;
    for (int i = t; i < K_CODES; i += 256) {
        unsigned h = hist[i];
        u += (h > 0) ? 1.f : 0.f;
        float p = (float)h * (1.0f / (float)N_VEC);
        H += p * logf(p + 1e-10f);
    }
    red[t] = u; __syncthreads();
    for (int k = 128; k > 0; k >>= 1) { if (t < k) red[t] += red[t + k]; __syncthreads(); }
    float util = red[0]; __syncthreads();
    red[t] = H; __syncthreads();
    for (int k = 128; k > 0; k >>= 1) { if (t < k) red[t] += red[t + k]; __syncthreads(); }
    float Ht = red[0];

    if (t == 0) {
        out4[0] = 1.25f * total / (float)(N_VEC * C_DIM);  // loss = (BETA+ALPHA)*mean
        out4[1] = total / (float)N_VEC;                    // quant_error
        out4[2] = util / (float)K_CODES;                   // utilization
        out4[3] = expf(-Ht);                               // perplexity
    }
}

extern "C" void kernel_launch(void* const* d_in, const int* in_sizes, int n_in,
                              void* d_out, int out_size, void* d_ws, size_t ws_size,
                              hipStream_t stream) {
    const float* z     = (const float*)d_in[0];   // (16,256,32,32)
    const float* emb   = (const float*)d_in[1];   // (8192,256)
    const float* eprob = (const float*)d_in[2];   // (8192,)
    float* out = (float*)d_out;
    char*  ws  = (char*)d_ws;

    // workspace layout (bytes)
    _Float16* zf16 = (_Float16*)(ws + 0);                    //  8,388,608
    _Float16* ef16 = (_Float16*)(ws + 8388608);              //  4,194,304
    float* znorm   = (float*)(ws + 12582912);                //     65,536
    float* enorm   = (float*)(ws + 12648448);                //     32,768
    unsigned long long* rowmin = (unsigned long long*)(ws + 12681216);  // 131,072
    unsigned long long* colmax = (unsigned long long*)(ws + 12812288);  //  65,536
    unsigned* hist  = (unsigned*)(ws + 12877824);            //     32,768
    float* rowsumsq = (float*)(ws + 12910592);               //     65,536

    vq_init<<<64, 256, 0, stream>>>(rowmin, colmax, hist);
    vq_prep_z<<<N_VEC, 256, 0, stream>>>(z, zf16, znorm);
    vq_prep_e<<<K_CODES, 256, 0, stream>>>(emb, ef16, enorm);

    dim3 grid(K_CODES / BN, N_VEC / BM);
    vq_dist<<<grid, 256, 0, stream>>>(zf16, ef16, znorm, enorm, rowmin, colmax);

    // output layout: [z_q 4194304][loss,qerr,util,perp][new_emb 2097152][new_prob 8192]
    vq_rows<<<N_VEC, 256, 0, stream>>>(z, emb, rowmin, hist, rowsumsq, out + 0);
    vq_cols<<<K_CODES, 256, 0, stream>>>(z, emb, eprob, colmax, hist,
                                         out + 4194308, out + 6291460);
    vq_scalars<<<1, 256, 0, stream>>>(rowsumsq, hist, out + 4194304);
}
